// FastKV_42228118454472
// MI455X (gfx1250) — matrix operand order repaired and checked by reading, not compiled, not run on hardware
//
#include <hip/hip_runtime.h>

// ---------------------------------------------------------------------------
// FastKV linear attention for MI455X (gfx1250, wave32, WMMA bf16)
// B=4, T=2048, D_MODEL=1024, D_KV=512
// ---------------------------------------------------------------------------

#define BQ 4
#define TT 2048
#define DM 1024
#define DK 512
#define BT (BQ * TT)          // 8192 flattened rows
#define CH 64                 // chunk length for the recurrence
#define SLICE 32              // d_v rows of state M owned per workgroup

typedef __attribute__((ext_vector_type(16))) __bf16 v16bf;
typedef __attribute__((ext_vector_type(8)))  __bf16 v8bf;
typedef __attribute__((ext_vector_type(4)))  __bf16 v4bf;
typedef __attribute__((ext_vector_type(8)))  float  v8f;

// --- WMMA fragment helpers (wave32, 16x16x32 bf16 -> f32) -------------------
// A operand: matrix stored row-major [M, K] (K contiguous).
// Lane l holds row (row0 + l%16), K elems {kb..kb+7, kb+16..kb+23}, kb=8*(l/16).
__device__ __forceinline__ v16bf load_a_frag(const __bf16* base, int ld,
                                             int row0, int k0, int lane) {
  int r  = row0 + (lane & 15);
  int kb = k0 + 8 * (lane >> 4);
  const __bf16* p = base + r * ld + kb;
  v8bf lo = *(const v8bf*)p;
  v8bf hi = *(const v8bf*)(p + 16);
  v16bf a;
#pragma unroll
  for (int i = 0; i < 8; ++i) { a[i] = lo[i]; a[i + 8] = hi[i]; }
  return a;
}

// B operand: matrix stored as [N, K] (i.e. B^T row-major, K contiguous).
// Lane l holds column n = n0 + l%16, K elems {kb..kb+7, kb+16..kb+23}, kb=k0+8*(l/16).
__device__ __forceinline__ v16bf load_b_frag(const __bf16* base, int ld,
                                             int n0, int k0, int lane) {
  int n  = n0 + (lane & 15);
  int kb = k0 + 8 * (lane >> 4);
  const __bf16* p = base + n * ld + kb;
  v8bf lo = *(const v8bf*)p;
  v8bf hi = *(const v8bf*)(p + 16);
  v16bf b;
#pragma unroll
  for (int i = 0; i < 8; ++i) { b[i] = lo[i]; b[i + 8] = hi[i]; }
  return b;
}

__device__ __forceinline__ v8f wmma_bf16(v16bf a, v16bf b, v8f c) {
  return __builtin_amdgcn_wmma_f32_16x16x32_bf16(
      /*neg_a=*/false, a, /*neg_b=*/false, b,
      /*c_mod=*/(short)0, c, /*reuse_a=*/false, /*reuse_b=*/false);
}

// --- Kernel 0: f32 -> bf16 conversion (x and weights) -----------------------
__global__ void k_cvt_bf16(const float* __restrict__ in,
                           __bf16* __restrict__ out, int n4) {
  int i = blockIdx.x * blockDim.x + threadIdx.x;
  if (i >= n4) return;
  float4 f = ((const float4*)in)[i];
  v4bf o;
  o[0] = (__bf16)f.x; o[1] = (__bf16)f.y; o[2] = (__bf16)f.z; o[3] = (__bf16)f.w;
  ((v4bf*)out)[i] = o;
}

// --- Kernel 1: fused q/k/v projection ---------------------------------------
// One wave per 16x16 output tile; produces Q,K row-major and K^T,V^T layouts.
// 2-stage software pipeline: loads for step kk+1 overlap WMMAs of step kk.
__global__ __launch_bounds__(256) void k_qkv(
    const __bf16* __restrict__ xh, const __bf16* __restrict__ Wq,
    const __bf16* __restrict__ Wk, const __bf16* __restrict__ Wv,
    __bf16* __restrict__ Q, __bf16* __restrict__ Kr,
    __bf16* __restrict__ Kt, __bf16* __restrict__ Vt) {
  int lane = threadIdx.x & 31;
  int w    = blockIdx.x * 8 + (threadIdx.x >> 5);
  int mt   = w >> 5;                // 0..511 : row tile of [8192]
  int nt   = w & 31;                // 0..31  : col tile of [512]
  int row0 = mt * 16;
  int col0 = nt * 16;
  v8f aq = {}, ak = {}, av = {};

  v16bf a  = load_a_frag(xh, DM, row0, 0, lane);
  v16bf bq = load_b_frag(Wq, DM, col0, 0, lane);
  v16bf bk = load_b_frag(Wk, DM, col0, 0, lane);
  v16bf bv = load_b_frag(Wv, DM, col0, 0, lane);
  for (int kk = 1; kk < DM / 32; ++kk) {
    int k0 = kk * 32;
    v16bf a2  = load_a_frag(xh, DM, row0, k0, lane);
    v16bf bq2 = load_b_frag(Wq, DM, col0, k0, lane);
    v16bf bk2 = load_b_frag(Wk, DM, col0, k0, lane);
    v16bf bv2 = load_b_frag(Wv, DM, col0, k0, lane);
    aq = wmma_bf16(a, bq, aq);
    ak = wmma_bf16(a, bk, ak);
    av = wmma_bf16(a, bv, av);
    a = a2; bq = bq2; bk = bk2; bv = bv2;
  }
  aq = wmma_bf16(a, bq, aq);
  ak = wmma_bf16(a, bk, ak);
  av = wmma_bf16(a, bv, av);

  // D fragment: lane l holds column n = l%16, rows m = 8*(l/16) + r.
  int n  = lane & 15;
  int mb = 8 * (lane >> 4);
  int d  = col0 + n;
  int b  = row0 >> 11;              // row tile never crosses batch (2048 % 16 == 0)
  int tb = (row0 & (TT - 1)) + mb;  // starting t within batch for this lane
#pragma unroll
  for (int r = 0; r < 8; ++r) {
    int g = row0 + mb + r;
    Q[g * DK + d]  = (__bf16)aq[r];
    Kr[g * DK + d] = (__bf16)ak[r];
  }
  v8bf pk, pv;
#pragma unroll
  for (int r = 0; r < 8; ++r) { pk[r] = (__bf16)ak[r]; pv[r] = (__bf16)av[r]; }
  // transposed layouts: contiguous along t -> single 16B store per lane
  *(v8bf*)(Kt + ((b * DK + d) * TT + tb)) = pk;
  *(v8bf*)(Vt + ((b * DK + d) * TT + tb)) = pv;
}

// --- Kernel 2: chunked linear-attention recurrence --------------------------
// Grid: 4 batches x 16 d_v slices. The f32 state M_slice [SLICE x 512] lives
// in persistent WMMA accumulator registers (8 tiles/wave x 8 VGPRs); only its
// bf16 shadow (B-operand for phase B) lives in LDS, refreshed once per chunk.
// Chunk C=64:  S = strict_tril(Q K^T);  Y = Q Mh^T + S V;  Macc += V^T K.
__global__ __launch_bounds__(256) void k_recur(
    const __bf16* __restrict__ Q, const __bf16* __restrict__ Kr,
    const __bf16* __restrict__ Kt, const __bf16* __restrict__ Vt,
    __bf16* __restrict__ Y) {
  __shared__ __bf16 Mh[SLICE * DK];   // 32 KB bf16 state shadow (B operand)
  __shared__ __bf16 Sh[CH * CH];      //  8 KB masked S tile (A operand)

  int lane = threadIdx.x & 31;
  int wave = threadIdx.x >> 5;
  int b    = blockIdx.x >> 4;
  int sl   = blockIdx.x & 15;
  int dv0  = sl * SLICE;

  const __bf16* Qb  = Q  + b * TT * DK;
  const __bf16* Kb  = Kr + b * TT * DK;
  const __bf16* Ktb = Kt + b * DK * TT;
  const __bf16* Vtb = Vt + b * DK * TT + dv0 * TT;
  __bf16* Yb = Y + b * TT * DK + dv0;

  // Persistent f32 state: wave owns tiles slot = wave*8+s,
  // tm = slot>>5 (d_v tile 0..1), tj = slot&31 (d_k tile 0..31).
  v8f Macc[8];
#pragma unroll
  for (int s = 0; s < 8; ++s) Macc[s] = (v8f){};

  for (int i = threadIdx.x; i < SLICE * DK; i += 256) Mh[i] = (__bf16)0.0f;
  __syncthreads();

  int n  = lane & 15;
  int mb = 8 * (lane >> 4);

  for (int ch = 0; ch < TT / CH; ++ch) {
    int t0 = ch * CH;

    // -------- Phase A: S = strict-lower(Q K^T), 16 tiles, 2 per wave -------
#pragma unroll
    for (int s = 0; s < 2; ++s) {
      int slot = wave * 2 + s;
      int ti = slot >> 2, tj = slot & 3;
      v8f acc = {};
      if (tj <= ti) {
        v16bf a  = load_a_frag(Qb, DK, t0 + ti * 16, 0, lane);
        v16bf bb = load_b_frag(Kb, DK, t0 + tj * 16, 0, lane);
        for (int kk = 1; kk < DK / 32; ++kk) {
          v16bf a2 = load_a_frag(Qb, DK, t0 + ti * 16, kk * 32, lane);
          v16bf b2 = load_b_frag(Kb, DK, t0 + tj * 16, kk * 32, lane);
          acc = wmma_bf16(a, bb, acc);
          a = a2; bb = b2;
        }
        acc = wmma_bf16(a, bb, acc);
      }
#pragma unroll
      for (int r = 0; r < 8; ++r) {
        int m = mb + r;
        bool keep = (tj * 16 + n) < (ti * 16 + m);   // strictly causal (s < t)
        Sh[(ti * 16 + m) * CH + tj * 16 + n] = keep ? (__bf16)acc[r] : (__bf16)0.0f;
      }
    }
    __syncthreads();

    // -------- Phase B: Y = Q Mh^T + S V ; 4x2 tiles, one per wave ----------
    {
      int tm = wave >> 1, tn = wave & 1;
      v8f acc = {};
      {
        v16bf a  = load_a_frag(Qb, DK, t0 + tm * 16, 0, lane);
        v16bf bb = load_b_frag(Mh, DK, tn * 16, 0, lane);
        for (int kk = 1; kk < DK / 32; ++kk) {        // inter-chunk: Q * Mh^T
          v16bf a2 = load_a_frag(Qb, DK, t0 + tm * 16, kk * 32, lane);
          v16bf b2 = load_b_frag(Mh, DK, tn * 16, kk * 32, lane);
          acc = wmma_bf16(a, bb, acc);
          a = a2; bb = b2;
        }
        acc = wmma_bf16(a, bb, acc);
      }
#pragma unroll
      for (int kk = 0; kk < CH / 32; ++kk) {          // intra-chunk: S * V
        v16bf a  = load_a_frag(Sh, CH, tm * 16, kk * 32, lane);
        v16bf bb = load_b_frag(Vtb, TT, tn * 16, t0 + kk * 32, lane);
        acc = wmma_bf16(a, bb, acc);
      }
#pragma unroll
      for (int r = 0; r < 8; ++r)
        Yb[(t0 + tm * 16 + mb + r) * DK + tn * 16 + n] = (__bf16)acc[r];
    }
    __syncthreads();

    // -------- Phase C: Macc += V^T K ; then refresh bf16 shadow ------------
#pragma unroll
    for (int s = 0; s < 8; ++s) {
      int slot = wave * 8 + s;
      int tm = slot >> 5;         // 0..1  (d_v tile within slice)
      int tj = slot & 31;         // 0..31 (d_k tile)
#pragma unroll
      for (int kk = 0; kk < CH / 32; ++kk) {
        v16bf a  = load_a_frag(Vtb, TT, tm * 16, t0 + kk * 32, lane);
        v16bf bb = load_b_frag(Ktb, TT, tj * 16, t0 + kk * 32, lane);
        Macc[s] = wmma_bf16(a, bb, Macc[s]);
      }
#pragma unroll
      for (int r = 0; r < 8; ++r)
        Mh[(tm * 16 + mb + r) * DK + tj * 16 + n] = (__bf16)Macc[s][r];
    }
    __syncthreads();
  }
}

// --- Kernel 3: output projection out = Y @ Wo^T (f32 result) ----------------
__global__ __launch_bounds__(256) void k_out(
    const __bf16* __restrict__ Y, const __bf16* __restrict__ Wo,
    float* __restrict__ out) {
  int lane = threadIdx.x & 31;
  int w    = blockIdx.x * 8 + (threadIdx.x >> 5);
  int mt   = w >> 6;               // 0..511
  int nt   = w & 63;               // 0..63
  v8f acc = {};
  v16bf a  = load_a_frag(Y,  DK, mt * 16, 0, lane);
  v16bf bb = load_b_frag(Wo, DK, nt * 16, 0, lane);
  for (int kk = 1; kk < DK / 32; ++kk) {
    v16bf a2 = load_a_frag(Y,  DK, mt * 16, kk * 32, lane);
    v16bf b2 = load_b_frag(Wo, DK, nt * 16, kk * 32, lane);
    acc = wmma_bf16(a, bb, acc);
    a = a2; bb = b2;
  }
  acc = wmma_bf16(a, bb, acc);
  int n  = lane & 15;
  int mb = 8 * (lane >> 4);
#pragma unroll
  for (int r = 0; r < 8; ++r)
    out[(mt * 16 + mb + r) * DM + nt * 16 + n] = acc[r];
}

// ---------------------------------------------------------------------------
extern "C" void kernel_launch(void* const* d_in, const int* in_sizes, int n_in,
                              void* d_out, int out_size, void* d_ws, size_t ws_size,
                              hipStream_t stream) {
  (void)in_sizes; (void)n_in; (void)out_size; (void)ws_size;
  const float* x  = (const float*)d_in[0];
  const float* Wq = (const float*)d_in[1];
  const float* Wk = (const float*)d_in[2];
  const float* Wv = (const float*)d_in[3];
  const float* Wo = (const float*)d_in[4];
  float* out = (float*)d_out;

  char* ws = (char*)d_ws;
  size_t off = 0;
  auto alloc = [&](size_t bytes) -> void* {
    off = (off + 255) & ~(size_t)255;
    void* p = ws + off;
    off += bytes;
    return p;
  };
  __bf16* xh  = (__bf16*)alloc((size_t)BT * DM * 2);   // 16 MB
  __bf16* Wqh = (__bf16*)alloc((size_t)DK * DM * 2);
  __bf16* Wkh = (__bf16*)alloc((size_t)DK * DM * 2);
  __bf16* Wvh = (__bf16*)alloc((size_t)DK * DM * 2);
  __bf16* Woh = (__bf16*)alloc((size_t)DM * DK * 2);
  __bf16* Q   = (__bf16*)alloc((size_t)BT * DK * 2);   // 8 MB each
  __bf16* Kr  = (__bf16*)alloc((size_t)BT * DK * 2);
  __bf16* Kt  = (__bf16*)alloc((size_t)BT * DK * 2);
  __bf16* Vt  = (__bf16*)alloc((size_t)BT * DK * 2);
  __bf16* Y   = (__bf16*)alloc((size_t)BT * DK * 2);

  // f32 -> bf16 conversions
  {
    int n4 = BT * DM / 4;
    k_cvt_bf16<<<(n4 + 255) / 256, 256, 0, stream>>>(x, xh, n4);
    int w4 = DK * DM / 4;
    k_cvt_bf16<<<(w4 + 255) / 256, 256, 0, stream>>>(Wq, Wqh, w4);
    k_cvt_bf16<<<(w4 + 255) / 256, 256, 0, stream>>>(Wk, Wkh, w4);
    k_cvt_bf16<<<(w4 + 255) / 256, 256, 0, stream>>>(Wv, Wvh, w4);
    k_cvt_bf16<<<(w4 + 255) / 256, 256, 0, stream>>>(Wo, Woh, w4);
  }
  // q/k/v projections: (8192/16)*(512/16) = 16384 waves -> 2048 blocks
  k_qkv<<<2048, 256, 0, stream>>>(xh, Wqh, Wkh, Wvh, Q, Kr, Kt, Vt);
  // recurrence: 4 batches x 16 slices
  k_recur<<<BQ * (DK / SLICE), 256, 0, stream>>>(Q, Kr, Kt, Vt, Y);
  // output projection: (8192/16)*(1024/16) = 32768 waves -> 4096 blocks
  k_out<<<4096, 256, 0, stream>>>(Y, Woh, out);
}
